// CameraAwareLoss_10814727651466
// MI455X (gfx1250) — compile-verified, matrix operand order checked by reading, not run on hardware
//
#include <hip/hip_runtime.h>
#include <hip/hip_bf16.h>

// ---------------------------------------------------------------------------
// CameraAwareLoss for MI455X (gfx1250, wave32, WMMA).
//   n_groups=4096, g=4, d=1024, batch=16384.
// Heavy part: argmax_j!=samelabel of fc_i . fc_j  (4096x4096x1024 gram),
// done with v_wmma_f32_16x16x32_bf16, fused with the masked argmax so the
// 64MB dist matrix is never materialized. Small per-group grams in VALU.
//
// Round-2 fix: round 1 showed LICM hoisting all k-steps of the (t0-loop
// invariant) A fragments out of the N-tile loop, spilling ~300B/lane to
// scratch (scratch_load_b128 reloads feeding the WMMAs). We now launder a
// zero byte-offset through an empty asm block inside each t0 iteration so
// the LDS loads cannot be hoisted/CSE'd across iterations, while the base
// pointer still provably derives from __shared__ (ds_load_b128 preserved).
// ---------------------------------------------------------------------------

#define D        1024
#define DPAD     (D + 8)      // LDS row stride in bf16 elements (2064 bytes)
#define GSZ      4
#define LMARGIN  0.3f

typedef __attribute__((ext_vector_type(16))) __bf16 v16bf;
typedef __attribute__((ext_vector_type(8)))  __bf16 v8bf;
typedef __attribute__((ext_vector_type(8)))  float  v8f;

__device__ __forceinline__ float dot4(const float4& a, const float4& b) {
    return a.x * b.x + a.y * b.y + a.z * b.z + a.w * b.w;
}

__device__ __forceinline__ float waveReduceSum(float v) {
#pragma unroll
    for (int off = 16; off > 0; off >>= 1) v += __shfl_xor(v, off, 32);
    return v;
}

// Load one 16-element bf16 fragment (row-major source, ISA 7.12.2 layout:
// per lane, K = ko..ko+7 and ko+16..ko+23). Works for LDS or global pointers.
__device__ __forceinline__ v16bf loadFrag(const __bf16* row, int ko) {
    v8bf lo = *(const v8bf*)(row + ko);
    v8bf hi = *(const v8bf*)(row + ko + 16);
    return __builtin_shufflevector(lo, hi, 0, 1, 2, 3, 4, 5, 6, 7,
                                   8, 9, 10, 11, 12, 13, 14, 15);
}

// ---------------------------------------------------------------------------
__global__ void init_out_kernel(float* __restrict__ out) {
    if (threadIdx.x == 0) out[0] = 0.0f;
}

// ---------------------------------------------------------------------------
// Per group: fc = l2norm(mean of 4 rows) -> bf16; inv_norm of each raw row.
__global__ __launch_bounds__(256)
void fc_norm_kernel(const float* __restrict__ inp,
                    float* __restrict__ inv_norm,
                    void* __restrict__ fcb_v) {
    __bf16* fcb = (__bf16*)fcb_v;
    const int g   = blockIdx.x;
    const int tid = threadIdx.x;

    const float4* base = (const float4*)(inp + (size_t)g * GSZ * D);
    float4 x0 = base[0 * (D / 4) + tid];
    float4 x1 = base[1 * (D / 4) + tid];
    float4 x2 = base[2 * (D / 4) + tid];
    float4 x3 = base[3 * (D / 4) + tid];

    float4 m;
    m.x = 0.25f * (x0.x + x1.x + x2.x + x3.x);
    m.y = 0.25f * (x0.y + x1.y + x2.y + x3.y);
    m.z = 0.25f * (x0.z + x1.z + x2.z + x3.z);
    m.w = 0.25f * (x0.w + x1.w + x2.w + x3.w);

    float vals[5];
    vals[0] = dot4(x0, x0);
    vals[1] = dot4(x1, x1);
    vals[2] = dot4(x2, x2);
    vals[3] = dot4(x3, x3);
    vals[4] = dot4(m, m);

    __shared__ float sacc[5];
    if (tid < 5) sacc[tid] = 0.0f;
    __syncthreads();
#pragma unroll
    for (int q = 0; q < 5; ++q) {
        float v = waveReduceSum(vals[q]);
        if ((tid & 31) == 0) atomicAdd(&sacc[q], v);
    }
    __syncthreads();

    if (tid < GSZ) inv_norm[g * GSZ + tid] = rsqrtf(sacc[tid]);

    const float invm = rsqrtf(sacc[4]);
    __bf16* frow = fcb + (size_t)g * D + tid * 4;
    frow[0] = (__bf16)(m.x * invm);
    frow[1] = (__bf16)(m.y * invm);
    frow[2] = (__bf16)(m.z * invm);
    frow[3] = (__bf16)(m.w * invm);
}

// ---------------------------------------------------------------------------
// Fused gram + masked argmax:
//   idx[i] = argmax_{j : label[j] != label[i]} fc_i . fc_j  (first index wins)
// One block per 16-row M tile (512 threads = 16 waves). A tile staged in LDS
// (padded rows). Each wave owns 8 pairs of adjacent N tiles; per k-step one
// A fragment feeds two v_wmma_f32_16x16x32_bf16.
__global__ __launch_bounds__(512)
void argmax_kernel(const void* __restrict__ fcb_v,
                   const int* __restrict__ target,
                   int* __restrict__ idx_out, int NG) {
    const __bf16* fcb = (const __bf16*)fcb_v;

    __shared__ __attribute__((aligned(16))) __bf16 As[16 * DPAD];  // ~32.3 KB
    __shared__ float sVal[16][16];
    __shared__ int   sIdx[16][16];

    const int tid    = threadIdx.x;
    const int m_base = blockIdx.x * 16;

    // Stage the 16 A rows into LDS with a 16B row pad (129 uint4 per row).
    {
        const uint4* src = (const uint4*)(fcb + (size_t)m_base * D);
        uint4*       dst = (uint4*)As;
        for (int i = tid; i < 16 * (D / 8); i += blockDim.x) {
            const int r = i >> 7;          // i / 128
            const int c = i & 127;         // i % 128
            dst[r * (DPAD / 8) + c] = src[i];
        }
    }
    __syncthreads();

    const int wave = tid >> 5;
    const int lane = tid & 31;
    const int rl   = lane & 15;   // row within A tile / col within C tile
    const int hi   = lane >> 4;   // lane-half selects K sub-block / M half
    const int kofs = hi ? 8 : 0;

    // Labels of the 8 M rows this lane's C registers hold.
    int lbM[8];
#pragma unroll
    for (int v = 0; v < 8; ++v) lbM[v] = target[GSZ * (m_base + hi * 8 + v)];

    float bestVal[8];
    int   bestIdx[8];
#pragma unroll
    for (int v = 0; v < 8; ++v) { bestVal[v] = -3.0e38f; bestIdx[v] = 0x7FFFFFFF; }

    const __bf16* Arow   = As + (size_t)rl * DPAD;
    const int     nTiles = NG / 16;

    // Wave w handles tile pairs (2w, 2w+1), (2w+32, 2w+33), ...
    for (int t0 = 2 * wave; t0 < nTiles; t0 += 32) {
        const int j0 = t0 * 16 + rl;
        const int j1 = j0 + 16;
        const __bf16* B0 = fcb + (size_t)j0 * D;
        const __bf16* B1 = fcb + (size_t)j1 * D;

        // Prefetch next iteration's B rows into cache.
        if (t0 + 32 < nTiles) {
            __builtin_prefetch(B0 + 32 * 16 * D, 0, 1);
            __builtin_prefetch(B1 + 32 * 16 * D, 0, 1);
        }

        // Optimization barrier: zbar is "redefined" here every iteration, so
        // the A-fragment LDS loads below depend on it and can neither be
        // hoisted out of the t0 loop (the round-1 scratch-spill cause) nor
        // CSE'd across iterations. Ap still provably points into As, so the
        // loads stay ds_load_b128.
        int zbar = 0;
        asm volatile("" : "+v"(zbar));
        const __bf16* Ap = Arow + zbar;

        v8f c0 = {0.f, 0.f, 0.f, 0.f, 0.f, 0.f, 0.f, 0.f};
        v8f c1 = {0.f, 0.f, 0.f, 0.f, 0.f, 0.f, 0.f, 0.f};

        // Software pipeline: fragments for step k are loaded one step ahead.
        v16bf a  = loadFrag(Ap, kofs);
        v16bf b0 = loadFrag(B0, kofs);
        v16bf b1 = loadFrag(B1, kofs);
#pragma unroll 4
        for (int k = 32; k < D; k += 32) {
            const int ko = k + kofs;
            v16bf an  = loadFrag(Ap, ko);
            v16bf b0n = loadFrag(B0, ko);
            v16bf b1n = loadFrag(B1, ko);
            c0 = __builtin_amdgcn_wmma_f32_16x16x32_bf16(
                     false, a, false, b0, (short)0, c0, false, false);
            c1 = __builtin_amdgcn_wmma_f32_16x16x32_bf16(
                     false, a, false, b1, (short)0, c1, false, false);
            a = an; b0 = b0n; b1 = b1n;
        }
        c0 = __builtin_amdgcn_wmma_f32_16x16x32_bf16(
                 false, a, false, b0, (short)0, c0, false, false);
        c1 = __builtin_amdgcn_wmma_f32_16x16x32_bf16(
                 false, a, false, b1, (short)0, c1, false, false);

        // Branchless masked argmax update (two columns per lane).
        const int lb0 = target[GSZ * j0];
        const int lb1 = target[GSZ * j1];
#pragma unroll
        for (int v = 0; v < 8; ++v) {
            float dv = c0[v];
            bool upd = (lbM[v] != lb0) &
                       ((dv > bestVal[v]) |
                        ((dv == bestVal[v]) & (j0 < bestIdx[v])));
            bestVal[v] = upd ? dv : bestVal[v];
            bestIdx[v] = upd ? j0 : bestIdx[v];

            dv  = c1[v];
            upd = (lbM[v] != lb1) &
                  ((dv > bestVal[v]) |
                   ((dv == bestVal[v]) & (j1 < bestIdx[v])));
            bestVal[v] = upd ? dv : bestVal[v];
            bestIdx[v] = upd ? j1 : bestIdx[v];
        }
    }

    // Reduce across the 16 lanes of each half-wave (they share the same rows).
#pragma unroll
    for (int off = 8; off > 0; off >>= 1) {
#pragma unroll
        for (int v = 0; v < 8; ++v) {
            const float ov  = __shfl_xor(bestVal[v], off, 32);
            const int   oi  = __shfl_xor(bestIdx[v], off, 32);
            const bool  upd = (ov > bestVal[v]) |
                              ((ov == bestVal[v]) & (oi < bestIdx[v]));
            bestVal[v] = upd ? ov : bestVal[v];
            bestIdx[v] = upd ? oi : bestIdx[v];
        }
    }
    if (rl == 0) {  // lanes 0 (rows 0..7) and 16 (rows 8..15)
#pragma unroll
        for (int v = 0; v < 8; ++v) {
            sVal[wave][hi * 8 + v] = bestVal[v];
            sIdx[wave][hi * 8 + v] = bestIdx[v];
        }
    }
    __syncthreads();

    if (tid < 16) {  // combine the 16 waves' disjoint N subsets
        float bv = -3.0e38f;
        int   bi = 0x7FFFFFFF;
        for (int w = 0; w < 16; ++w) {
            const float ov  = sVal[w][tid];
            const int   oi  = sIdx[w][tid];
            const bool  upd = (ov > bv) | ((ov == bv) & (oi < bi));
            bv = upd ? ov : bv;
            bi = upd ? oi : bi;
        }
        idx_out[m_base + tid] = (bi == 0x7FFFFFFF) ? 0 : bi;
    }
}

// ---------------------------------------------------------------------------
// Per-group 4x4 gram with camera mask -> m_a, c1.
__global__ __launch_bounds__(256)
void ma_kernel(const float* __restrict__ inp,
               const int* __restrict__ cam,
               const float* __restrict__ inv_norm,
               float* __restrict__ m_a,
               float* __restrict__ c1out) {
    const int g   = blockIdx.x;
    const int tid = threadIdx.x;

    float4 x[4];
#pragma unroll
    for (int r = 0; r < 4; ++r) {
        const float s = inv_norm[g * GSZ + r];
        float4 t = ((const float4*)(inp + (size_t)(g * GSZ + r) * D))[tid];
        x[r] = make_float4(t.x * s, t.y * s, t.z * s, t.w * s);
    }

    const int pi[6] = {0, 0, 0, 1, 1, 2};
    const int pj[6] = {1, 2, 3, 2, 3, 3};
    float acc[6];
#pragma unroll
    for (int q = 0; q < 6; ++q) acc[q] = dot4(x[pi[q]], x[pj[q]]);

    __shared__ float sacc[6];
    if (tid < 6) sacc[tid] = 0.0f;
    __syncthreads();
#pragma unroll
    for (int q = 0; q < 6; ++q) {
        const float v = waveReduceSum(acc[q]);
        if ((tid & 31) == 0) atomicAdd(&sacc[q], v);
    }
    __syncthreads();

    if (tid == 0) {
        int c[4];
#pragma unroll
        for (int r = 0; r < 4; ++r) c[r] = cam[g * GSZ + r];
        float s1 = 0.0f, c1 = 0.0f;
#pragma unroll
        for (int q = 0; q < 6; ++q) {
            if (c[pi[q]] != c[pj[q]]) { s1 += -2.0f * sacc[q]; c1 += 2.0f; }
        }
        m_a[g]   = s1 / fmaxf(c1, 1.0f);
        c1out[g] = c1;
    }
}

// ---------------------------------------------------------------------------
// Per-group 8x8 gram over [fa, fa[idx]] with (cam!=cam && tgt!=tgt) mask,
// combine with m_a / m_a[idx], relu-margin, atomic add into the scalar loss.
__global__ __launch_bounds__(256)
void loss_kernel(const float* __restrict__ inp,
                 const int* __restrict__ target,
                 const int* __restrict__ cam,
                 const float* __restrict__ inv_norm,
                 const int* __restrict__ idx,
                 const float* __restrict__ m_a,
                 const float* __restrict__ c1arr,
                 float* __restrict__ out) {
    const int g   = blockIdx.x;
    const int tid = threadIdx.x;
    const int jg  = idx[g];

    int rows[8];
#pragma unroll
    for (int p = 0; p < 8; ++p)
        rows[p] = (p < 4) ? (g * GSZ + p) : (jg * GSZ + (p - 4));

    float4 x[8];
#pragma unroll
    for (int p = 0; p < 8; ++p) {
        const float s = inv_norm[rows[p]];
        float4 t = ((const float4*)(inp + (size_t)rows[p] * D))[tid];
        x[p] = make_float4(t.x * s, t.y * s, t.z * s, t.w * s);
    }

    float acc[28];
    {
        int q = 0;
#pragma unroll
        for (int p = 0; p < 8; ++p)
#pragma unroll
            for (int r = p + 1; r < 8; ++r, ++q) acc[q] = dot4(x[p], x[r]);
    }

    __shared__ float sacc[28];
    if (tid < 28) sacc[tid] = 0.0f;
    __syncthreads();
#pragma unroll
    for (int q = 0; q < 28; ++q) {
        const float v = waveReduceSum(acc[q]);
        if ((tid & 31) == 0) atomicAdd(&sacc[q], v);
    }
    __syncthreads();

    if (tid == 0) {
        int cv[8], tv[8];
#pragma unroll
        for (int p = 0; p < 8; ++p) {
            cv[p] = cam[rows[p]];
            tv[p] = target[rows[p]];
        }
        float s2 = 0.0f, c2 = 0.0f;
        int q = 0;
#pragma unroll
        for (int p = 0; p < 8; ++p)
#pragma unroll
            for (int r = p + 1; r < 8; ++r, ++q)
                if (cv[p] != cv[r] && tv[p] != tv[r]) {
                    s2 += -2.0f * sacc[q];
                    c2 += 2.0f;
                }
        const float m2 = s2 / fmaxf(c2, 1.0f);
        const bool  v2 = c2 > 0.0f;
        const bool  va = c1arr[g] > 0.0f;
        const bool  vb = c1arr[jg] > 0.0f;
        const float ma = m_a[g];
        const float mb = m_a[jg];
        const float m1 = (va && vb) ? 0.5f * (ma + mb) : (va ? ma : mb);
        const bool  valid = (va || vb) && v2;
        const float loss = fmaxf(0.0f, m1 - m2 + LMARGIN);
        if (valid) atomicAdd(out, loss);
    }
}

// ---------------------------------------------------------------------------
extern "C" void kernel_launch(void* const* d_in, const int* in_sizes, int n_in,
                              void* d_out, int out_size, void* d_ws, size_t ws_size,
                              hipStream_t stream) {
    const float* inp    = (const float*)d_in[0];
    const int*   target = (const int*)d_in[1];
    const int*   cam    = (const int*)d_in[2];
    float*       out    = (float*)d_out;

    const int batch = in_sizes[0] / D;   // 16384
    const int NG    = batch / GSZ;       // 4096

    // Workspace layout (~8.1 MB): bf16 fc | row inv-norms | idx | m_a | c1
    char*  ws       = (char*)d_ws;
    size_t off      = 0;
    void*  fcb      = (void*)(ws + off);        off += (size_t)NG * D * 2;
    off = (off + 255) & ~(size_t)255;
    float* inv_norm = (float*)(ws + off);       off += (size_t)batch * sizeof(float);
    int*   idx      = (int*)(ws + off);         off += (size_t)NG * sizeof(int);
    float* m_a      = (float*)(ws + off);       off += (size_t)NG * sizeof(float);
    float* c1a      = (float*)(ws + off);       off += (size_t)NG * sizeof(float);
    (void)ws_size; (void)n_in; (void)out_size;

    init_out_kernel<<<1, 1, 0, stream>>>(out);
    fc_norm_kernel<<<NG, 256, 0, stream>>>(inp, inv_norm, fcb);
    argmax_kernel<<<NG / 16, 512, 0, stream>>>(fcb, target, idx, NG);
    ma_kernel<<<NG, 256, 0, stream>>>(inp, cam, inv_norm, m_a, c1a);
    loss_kernel<<<NG, 256, 0, stream>>>(inp, target, cam, inv_norm, idx, m_a, c1a, out);
}
